// SwinTransformerBlock_55310588838458
// MI455X (gfx1250) — compile-verified, matrix operand order checked by reading, not run on hardware
//
#include <hip/hip_runtime.h>
#include <hip/hip_bf16.h>
#include <stdint.h>

typedef __bf16 bf16;
typedef __attribute__((ext_vector_type(16))) __bf16 v16bf;
typedef __attribute__((ext_vector_type(8)))  float  v8f;
typedef __attribute__((ext_vector_type(4)))  int    v4i_;

#define WS_ 8
#define SS_ 4
#define NTOK 64           // tokens per window
#define CDIM 256
#define NH_ 8
#define HD_ 32
#define NWIN 64           // windows per image
#define BIMG 32
#define BWIN 2048         // BIMG * NWIN
#define ROWS 131072       // BWIN * NTOK
#define HIDDEN_ 1024

// gfx1250 async global->LDS path (ASYNCcnt-tracked), with safe fallback
#if defined(__gfx1250__) && __has_builtin(__builtin_amdgcn_global_load_async_to_lds_b128) && __has_builtin(__builtin_amdgcn_s_wait_asynccnt)
#define USE_ASYNC_LDS 1
#else
#define USE_ASYNC_LDS 0
#endif

union Frag16 { uint4 u[2]; v16bf v; };

__device__ __forceinline__ v8f wmma_bf16(v16bf a, v16bf b, v8f c) {
  return __builtin_amdgcn_wmma_f32_16x16x32_bf16(false, a, false, b, (short)0, c, false, false);
}

__device__ __forceinline__ float gelu_tanh(float x) {
  float x3 = x * x * x;
  return 0.5f * x * (1.f + tanhf(0.7978845608028654f * (x + 0.044715f * x3)));
}
__device__ __forceinline__ float sigmoidf_(float x) { return 1.f / (1.f + __expf(-x)); }
__device__ __forceinline__ int reg3_(int t) { return t < 56 ? 0 : (t < 60 ? 1 : 2); }

// ---------------- weight fp32 -> bf16 cast ----------------
__global__ __launch_bounds__(256) void cvt_kernel(const float* __restrict__ in,
                                                  bf16* __restrict__ out, int n) {
  int i = blockIdx.x * 256 + threadIdx.x;
  if (i < n) out[i] = (bf16)in[i];
}

// ---------------- CPB MLP: table (225 x NH) ----------------
__global__ __launch_bounds__(256) void cpb_kernel(const float* __restrict__ w1,
                                                  const float* __restrict__ b1,
                                                  const float* __restrict__ w2,
                                                  float* __restrict__ tab) {
  int p = threadIdx.x;
  if (p >= 225) return;
  int i = p / 15, j = p % 15;
  float t0 = (float)(i - 7) * (8.f / 7.f);
  float t1 = (float)(j - 7) * (8.f / 7.f);
  float a0 = fabsf(t0), a1 = fabsf(t1);
  t0 = (t0 > 0.f ? 1.f : (t0 < 0.f ? -1.f : 0.f)) * log2f(a0 + 1.f) / 3.0f;
  t1 = (t1 > 0.f ? 1.f : (t1 < 0.f ? -1.f : 0.f)) * log2f(a1 + 1.f) / 3.0f;
  float acc[NH_];
#pragma unroll
  for (int h = 0; h < NH_; ++h) acc[h] = 0.f;
  for (int k = 0; k < 512; ++k) {
    float hv = fmaf(t0, w1[k], fmaf(t1, w1[512 + k], b1[k]));
    hv = fmaxf(hv, 0.f);
#pragma unroll
    for (int h = 0; h < NH_; ++h) acc[h] = fmaf(hv, w2[k * NH_ + h], acc[h]);
  }
#pragma unroll
  for (int h = 0; h < NH_; ++h) tab[p * NH_ + h] = acc[h];
}

// ---------------- roll(-4,-4) + window partition + bf16 cast ----------------
__global__ __launch_bounds__(256) void winpart_kernel(const float* __restrict__ x,
                                                      bf16* __restrict__ xw) {
  int rowid = blockIdx.x;                // b_ * 64 + nn
  int b_ = rowid >> 6, nn = rowid & 63;
  int img = b_ >> 6, w = b_ & 63;
  int wh = w >> 3, ww = w & 7;
  int r = nn >> 3, c = nn & 7;
  int y  = (wh * 8 + r + SS_) & 63;
  int xx = (ww * 8 + c + SS_) & 63;
  float v = x[(((size_t)img * 4096) + (size_t)y * 64 + xx) * CDIM + threadIdx.x];
  xw[(size_t)rowid * CDIM + threadIdx.x] = (bf16)v;
}

// ---------------- generic bf16 WMMA GEMM, 64x128 block tile ----------------
// EPI: 0 = QKV (q/v bias, scatter to per-head layout)
//      1 = PROJ (bias, window-reverse + roll scatter, fp32 out)
//      2 = FC1  (bias + GELU, bf16 out)
//      3 = FC2  (bias, fp32 out)
template <int EPI>
__global__ __launch_bounds__(256) void gemm_kernel(const bf16* __restrict__ A,
                                                   const bf16* __restrict__ Bw,
                                                   int M, int K, int Nld,
                                                   const float* __restrict__ bias0,
                                                   const float* __restrict__ bias1,
                                                   float* __restrict__ outF,
                                                   bf16* __restrict__ outB) {
  __shared__ bf16 As[64][40];      // padded stride: 80B -> conflict-free frag reads
  __shared__ bf16 Bt[128][40];     // B tile transposed [n][k]
  const int n0 = blockIdx.x * 128;
  const int m0 = blockIdx.y * 64;
  const int tid = threadIdx.x;
  const int lane = tid & 31;
  const int wave = tid >> 5;
  const int wm = wave & 3;         // 16-row strip
  const int wn = wave >> 2;        // 64-col half
  const int rlane = lane & 15;
  const int half = lane >> 4;      // 0/1 -> K/M sub-block
  const int kbase = half * 16;

  v8f acc[4];
#pragma unroll
  for (int t = 0; t < 4; ++t) acc[t] = (v8f){0.f,0.f,0.f,0.f,0.f,0.f,0.f,0.f};

  const int arow = tid >> 2;             // 0..63
  const int ako  = (tid & 3) * 8;        // 0,8,16,24
  const int bkr  = tid >> 3;             // 0..31
  const int bno  = (tid & 7) * 16;       // 0..112

  for (int kk = 0; kk < K; kk += 32) {
    // stage A tile 64x32 (bf16), 16B per thread: async DMA to LDS on gfx1250
    {
      const bf16* src = A + (size_t)(m0 + arow) * K + kk + ako;
#if USE_ASYNC_LDS
      __builtin_amdgcn_global_load_async_to_lds_b128(
          (__attribute__((address_space(1))) v4i_*)(uintptr_t)src,
          (__attribute__((address_space(3))) v4i_*)(uint32_t)(uintptr_t)&As[arow][ako],
          0, 0);
#else
      *(uint4*)&As[arow][ako] = *(const uint4*)src;
#endif
      if (kk + 32 < K) __builtin_prefetch(src + 32, 0, 3);   // global_prefetch_b8
    }
    // stage B tile 32x128 transposed, 32B per thread
    {
      const bf16* src = Bw + (size_t)(kk + bkr) * Nld + n0 + bno;
      uint4 d0 = *(const uint4*)src;
      uint4 d1 = *(const uint4*)(src + 8);
      bf16 tmp[16];
      *(uint4*)&tmp[0] = d0;
      *(uint4*)&tmp[8] = d1;
#pragma unroll
      for (int i = 0; i < 16; ++i) Bt[bno + i][bkr] = tmp[i];
    }
#if USE_ASYNC_LDS
    __builtin_amdgcn_s_wait_asynccnt(0);   // own async writes landed before barrier
#endif
    __syncthreads();
    Frag16 af;
    af.u[0] = *(const uint4*)&As[wm * 16 + rlane][kbase];
    af.u[1] = *(const uint4*)&As[wm * 16 + rlane][kbase + 8];
#pragma unroll
    for (int t = 0; t < 4; ++t) {
      int col = wn * 64 + t * 16 + rlane;
      Frag16 bf;
      bf.u[0] = *(const uint4*)&Bt[col][kbase];
      bf.u[1] = *(const uint4*)&Bt[col][kbase + 8];
      acc[t] = wmma_bf16(af.v, bf.v, acc[t]);
    }
    __syncthreads();
  }

  // epilogue
#pragma unroll
  for (int t = 0; t < 4; ++t) {
#pragma unroll
    for (int j = 0; j < 8; ++j) {
      int m = m0 + wm * 16 + half * 8 + j;
      int n = n0 + wn * 64 + t * 16 + rlane;
      float v = acc[t][j];
      if (EPI == 0) {
        int which = n >> 8, cc = n & 255;
        if (which == 0) v += bias0[cc];
        if (which == 2) v += bias1[cc];
        int head = cc >> 5, d = cc & 31;
        int b_ = m >> 6, nn = m & 63;
        size_t dst = (((size_t)which * (BWIN * NH_) + (size_t)b_ * NH_ + head) * NTOK + nn) * HD_ + d;
        outB[dst] = (bf16)v;
      } else if (EPI == 1) {
        v += bias0[n];
        int b_ = m >> 6, nn = m & 63;
        int img = b_ >> 6, w = b_ & 63;
        int wh = w >> 3, ww = w & 7;
        int r = nn >> 3, c = nn & 7;
        int y  = (wh * 8 + r + SS_) & 63;
        int xx = (ww * 8 + c + SS_) & 63;
        outF[(((size_t)img * 4096) + (size_t)y * 64 + xx) * CDIM + n] = v;
      } else if (EPI == 2) {
        v = gelu_tanh(v + bias0[n]);
        outB[(size_t)m * Nld + n] = (bf16)v;
      } else {
        v += bias0[n];
        outF[(size_t)m * Nld + n] = v;
      }
    }
  }
}

// ---------------- attention: one block per (window, head), 4 waves ----------------
__global__ __launch_bounds__(128) void attn_kernel(const bf16* __restrict__ qkv,
                                                   const float* __restrict__ logit_scale,
                                                   const float* __restrict__ tab,
                                                   bf16* __restrict__ attn_out) {
  __shared__ bf16 qn[64][40];
  __shared__ bf16 kn[64][40];
  __shared__ bf16 vt[32][72];    // v transposed [d][n]
  __shared__ float S[64][65];
  __shared__ bf16 P[64][72];

  const int blk = blockIdx.x;          // b_ * NH + h
  const int b_ = blk >> 3;
  const int h  = blk & 7;
  const int tid = threadIdx.x;
  const int lane = tid & 31;
  const int wv = tid >> 5;             // wave 0..3 -> 16-row strip
  const int rlane = lane & 15;
  const int half = lane >> 4;
  const int kbase = half * 16;

  const size_t qoff = ((size_t)0 * (BWIN * NH_) + (size_t)b_ * NH_ + h) * (NTOK * HD_);
  const size_t koff = ((size_t)1 * (BWIN * NH_) + (size_t)b_ * NH_ + h) * (NTOK * HD_);
  const size_t voff = ((size_t)2 * (BWIN * NH_) + (size_t)b_ * NH_ + h) * (NTOK * HD_);

  // phase 1: L2-normalize q,k rows; stage v transposed
  if (tid < 64) {
    const bf16* src = qkv + qoff + (size_t)tid * HD_;
    float row[HD_], ss = 0.f;
#pragma unroll
    for (int d = 0; d < HD_; ++d) { row[d] = (float)src[d]; ss = fmaf(row[d], row[d], ss); }
    float rn = rsqrtf(fmaxf(ss, 1e-12f));
#pragma unroll
    for (int d = 0; d < HD_; ++d) qn[tid][d] = (bf16)(row[d] * rn);
  } else {
    int rr = tid - 64;
    const bf16* src = qkv + koff + (size_t)rr * HD_;
    float row[HD_], ss = 0.f;
#pragma unroll
    for (int d = 0; d < HD_; ++d) { row[d] = (float)src[d]; ss = fmaf(row[d], row[d], ss); }
    float rn = rsqrtf(fmaxf(ss, 1e-12f));
#pragma unroll
    for (int d = 0; d < HD_; ++d) kn[rr][d] = (bf16)(row[d] * rn);
  }
  {
    int nn = tid >> 1, d0 = (tid & 1) * 16;
    const bf16* src = qkv + voff + (size_t)nn * HD_ + d0;
#pragma unroll
    for (int i = 0; i < 16; ++i) vt[d0 + i][nn] = src[i];
  }
  __syncthreads();

  // phase 2: S = (qn . kn^T) * scale + 16*sigmoid(cpb) + shift-mask
  const float scale = __expf(fminf(logit_scale[h], 4.605170185988091f)); // ln(100)
  const int w = b_ & 63, wh = w >> 3, ww = w & 7;
  {
    Frag16 af;
    af.u[0] = *(const uint4*)&qn[wv * 16 + rlane][kbase];
    af.u[1] = *(const uint4*)&qn[wv * 16 + rlane][kbase + 8];
    v8f sa[4];
#pragma unroll
    for (int t = 0; t < 4; ++t) sa[t] = (v8f){0.f,0.f,0.f,0.f,0.f,0.f,0.f,0.f};
#pragma unroll
    for (int t = 0; t < 4; ++t) {
      Frag16 bfr;
      bfr.u[0] = *(const uint4*)&kn[t * 16 + rlane][kbase];
      bfr.u[1] = *(const uint4*)&kn[t * 16 + rlane][kbase + 8];
      sa[t] = wmma_bf16(af.v, bfr.v, sa[t]);
    }
    const bool edge = (wh == 7) || (ww == 7);
#pragma unroll
    for (int t = 0; t < 4; ++t) {
#pragma unroll
      for (int j = 0; j < 8; ++j) {
        int mi = wv * 16 + half * 8 + j;
        int ni = t * 16 + rlane;
        int ri = mi >> 3, ci = mi & 7, rj = ni >> 3, cj = ni & 7;
        int idx = (ri - rj + 7) * 15 + (ci - cj + 7);
        float val = sa[t][j] * scale + 16.f * sigmoidf_(tab[idx * NH_ + h]);
        if (edge) {
          int regi = reg3_(wh * 8 + ri) * 3 + reg3_(ww * 8 + ci);
          int regj = reg3_(wh * 8 + rj) * 3 + reg3_(ww * 8 + cj);
          if (regi != regj) val += -100.f;
        }
        S[mi][ni] = val;
      }
    }
  }
  __syncthreads();

  // phase 3: softmax (one thread per row; exp stored back to S -> no reg array)
  if (tid < 64) {
    float mx = -1e30f;
#pragma unroll
    for (int i = 0; i < 64; ++i) mx = fmaxf(mx, S[tid][i]);
    float sum = 0.f;
#pragma unroll
    for (int i = 0; i < 64; ++i) { float e = __expf(S[tid][i] - mx); S[tid][i] = e; sum += e; }
    float inv = 1.f / sum;
#pragma unroll
    for (int i = 0; i < 64; ++i) P[tid][i] = (bf16)(S[tid][i] * inv);
  }
  __syncthreads();

  // phase 4: O = P @ V  (M=64, K=64, N=32)
  {
    v8f oa[2];
#pragma unroll
    for (int t = 0; t < 2; ++t) oa[t] = (v8f){0.f,0.f,0.f,0.f,0.f,0.f,0.f,0.f};
#pragma unroll
    for (int k2 = 0; k2 < 2; ++k2) {
      Frag16 af;
      af.u[0] = *(const uint4*)&P[wv * 16 + rlane][k2 * 32 + kbase];
      af.u[1] = *(const uint4*)&P[wv * 16 + rlane][k2 * 32 + kbase + 8];
#pragma unroll
      for (int t = 0; t < 2; ++t) {
        Frag16 bfr;
        bfr.u[0] = *(const uint4*)&vt[t * 16 + rlane][k2 * 32 + kbase];
        bfr.u[1] = *(const uint4*)&vt[t * 16 + rlane][k2 * 32 + kbase + 8];
        oa[t] = wmma_bf16(af.v, bfr.v, oa[t]);
      }
    }
#pragma unroll
    for (int t = 0; t < 2; ++t) {
#pragma unroll
      for (int j = 0; j < 8; ++j) {
        int mi = wv * 16 + half * 8 + j;
        int ni = t * 16 + rlane;
        attn_out[((size_t)b_ * NTOK + mi) * CDIM + h * HD_ + ni] = (bf16)oa[t][j];
      }
    }
  }
}

// ---------------- LayerNorm + residual row kernels ----------------
__global__ __launch_bounds__(256) void ln1_kernel(const float* __restrict__ xin,
                                                  const float* __restrict__ pout,
                                                  const float* __restrict__ sc,
                                                  const float* __restrict__ bi,
                                                  float* __restrict__ x1f,
                                                  bf16* __restrict__ x1b) {
  __shared__ float red[256];
  size_t row = blockIdx.x;
  int tid = threadIdx.x;
  float y = pout[row * CDIM + tid];
  red[tid] = y; __syncthreads();
  for (int s = 128; s > 0; s >>= 1) { if (tid < s) red[tid] += red[tid + s]; __syncthreads(); }
  float mu = red[0] * (1.f / CDIM); __syncthreads();
  float d = y - mu;
  red[tid] = d * d; __syncthreads();
  for (int s = 128; s > 0; s >>= 1) { if (tid < s) red[tid] += red[tid + s]; __syncthreads(); }
  float var = red[0] * (1.f / CDIM);
  float ln = d * rsqrtf(var + 1e-6f) * sc[tid] + bi[tid];
  float v = xin[row * CDIM + tid] + ln;
  x1f[row * CDIM + tid] = v;
  x1b[row * CDIM + tid] = (bf16)v;
}

__global__ __launch_bounds__(256) void ln2_kernel(const float* __restrict__ x1f,
                                                  const float* __restrict__ mout,
                                                  const float* __restrict__ sc,
                                                  const float* __restrict__ bi,
                                                  float* __restrict__ out) {
  __shared__ float red[256];
  size_t row = blockIdx.x;
  int tid = threadIdx.x;
  float y = mout[row * CDIM + tid];
  red[tid] = y; __syncthreads();
  for (int s = 128; s > 0; s >>= 1) { if (tid < s) red[tid] += red[tid + s]; __syncthreads(); }
  float mu = red[0] * (1.f / CDIM); __syncthreads();
  float d = y - mu;
  red[tid] = d * d; __syncthreads();
  for (int s = 128; s > 0; s >>= 1) { if (tid < s) red[tid] += red[tid + s]; __syncthreads(); }
  float var = red[0] * (1.f / CDIM);
  float ln = d * rsqrtf(var + 1e-6f) * sc[tid] + bi[tid];
  out[row * CDIM + tid] = x1f[row * CDIM + tid] + ln;
}

// ---------------- host launch ----------------
extern "C" void kernel_launch(void* const* d_in, const int* in_sizes, int n_in,
                              void* d_out, int out_size, void* d_ws, size_t ws_size,
                              hipStream_t stream) {
  const float* x          = (const float*)d_in[0];
  const float* qkv_w      = (const float*)d_in[1];
  const float* q_bias     = (const float*)d_in[2];
  const float* v_bias     = (const float*)d_in[3];
  const float* logitScale = (const float*)d_in[4];
  const float* cpb_w1     = (const float*)d_in[5];
  const float* cpb_b1     = (const float*)d_in[6];
  const float* cpb_w2     = (const float*)d_in[7];
  const float* proj_w     = (const float*)d_in[8];
  const float* proj_b     = (const float*)d_in[9];
  const float* n1s        = (const float*)d_in[10];
  const float* n1b        = (const float*)d_in[11];
  const float* n2s        = (const float*)d_in[12];
  const float* n2b        = (const float*)d_in[13];
  const float* fc1_w      = (const float*)d_in[14];
  const float* fc1_b      = (const float*)d_in[15];
  const float* fc2_w      = (const float*)d_in[16];
  const float* fc2_b      = (const float*)d_in[17];
  float* out = (float*)d_out;

  char* ws = (char*)d_ws;
  // region A (67MB): xw, later reused for attn_out
  bf16*  xw       = (bf16*)(ws + 0);
  bf16*  attn_out = (bf16*)(ws + 0);
  // region B (201MB): qkv, later reused for proj_out / mlp_out
  bf16*  qkv      = (bf16*)(ws + (size_t)67108864);
  float* proj_out = (float*)(ws + (size_t)67108864);
  float* mlp_out  = (float*)(ws + (size_t)67108864);
  bf16*  x1b      = (bf16*)(ws + (size_t)67108864 + 134217728);  // tail of region B
  // region C: x1 fp32
  float* x1f      = (float*)(ws + (size_t)268435456);
  // region D: hidden (bf16, 268MB)
  bf16*  hbuf     = (bf16*)(ws + (size_t)402653184);
  // weights (bf16) + cpb table
  size_t woff = (size_t)402653184 + 268435456;
  bf16*  wqkv = (bf16*)(ws + woff);              // 196608
  bf16*  wproj = (bf16*)(ws + woff + 524288);    // 65536
  bf16*  wfc1 = (bf16*)(ws + woff + 1048576);    // 262144
  bf16*  wfc2 = (bf16*)(ws + woff + 1572864);    // 262144
  float* tab  = (float*)(ws + woff + 2097152);   // 225*8 floats
  (void)ws_size; (void)in_sizes; (void)n_in; (void)out_size;

  // weight casts + CPB table
  cvt_kernel<<<768, 256, 0, stream>>>(qkv_w, wqkv, 196608);
  cvt_kernel<<<256, 256, 0, stream>>>(proj_w, wproj, 65536);
  cvt_kernel<<<1024, 256, 0, stream>>>(fc1_w, wfc1, 262144);
  cvt_kernel<<<1024, 256, 0, stream>>>(fc2_w, wfc2, 262144);
  cpb_kernel<<<1, 256, 0, stream>>>(cpb_w1, cpb_b1, cpb_w2, tab);

  // roll + window partition + bf16
  winpart_kernel<<<ROWS, 256, 0, stream>>>(x, xw);

  // QKV GEMM: (131072 x 256) @ (256 x 768)
  gemm_kernel<0><<<dim3(768 / 128, ROWS / 64), 256, 0, stream>>>(
      xw, wqkv, ROWS, CDIM, 768, q_bias, v_bias, nullptr, qkv);

  // attention per (window, head)
  attn_kernel<<<BWIN * NH_, 128, 0, stream>>>(qkv, logitScale, tab, attn_out);

  // proj GEMM with window-reverse + roll scatter
  gemm_kernel<1><<<dim3(CDIM / 128, ROWS / 64), 256, 0, stream>>>(
      attn_out, wproj, ROWS, CDIM, CDIM, proj_b, nullptr, proj_out, nullptr);

  // x1 = shortcut + LN1(proj_out)
  ln1_kernel<<<ROWS, 256, 0, stream>>>(x, proj_out, n1s, n1b, x1f, x1b);

  // MLP
  gemm_kernel<2><<<dim3(HIDDEN_ / 128, ROWS / 64), 256, 0, stream>>>(
      x1b, wfc1, ROWS, CDIM, HIDDEN_, fc1_b, nullptr, nullptr, hbuf);
  gemm_kernel<3><<<dim3(CDIM / 128, ROWS / 64), 256, 0, stream>>>(
      hbuf, wfc2, ROWS, HIDDEN_, CDIM, fc2_b, nullptr, mlp_out, nullptr);

  // out = x1 + LN2(mlp)
  ln2_kernel<<<ROWS, 256, 0, stream>>>(x1f, mlp_out, n2s, n2b, out);
}
